// TransformerBlock_12128987644300
// MI455X (gfx1250) — matrix operand order repaired and checked by reading, not compiled, not run on hardware
//
#include <hip/hip_runtime.h>

// ---------------------------------------------------------------------------
// Types for CDNA5 WMMA (wave32, 16x16x32 bf16 -> f32)
// ---------------------------------------------------------------------------
typedef __bf16 bf16_t;
typedef bf16_t bf16x4  __attribute__((ext_vector_type(4)));
typedef bf16_t bf16x8  __attribute__((ext_vector_type(8)));
typedef bf16_t bf16x16 __attribute__((ext_vector_type(16)));
typedef float  f32x8   __attribute__((ext_vector_type(8)));

#define DEV __device__ __forceinline__

// ---- CDNA5 async global->LDS copy (ASYNCcnt), with safe fallback ----------
// clang signature: (int4 AS1*, int4 AS3*, imm offset, imm cpol); the ISA adds
// INST_OFFSET to BOTH the global and the LDS address, so contiguous runs can
// share one address pair across several copies.
#if __has_builtin(__builtin_amdgcn_global_load_async_to_lds_b128)
#define HAVE_ASYNC 1
typedef int v4i_t __attribute__((vector_size(16)));
typedef __attribute__((address_space(1))) v4i_t* as1_v4i_p;
typedef __attribute__((address_space(3))) v4i_t* as3_v4i_p;
#else
#define HAVE_ASYNC 0
#endif

// copy 32 contiguous bytes (2 x b128)
static DEV void async_cp32B(const bf16_t* g, bf16_t* l) {
#if HAVE_ASYNC
    __builtin_amdgcn_global_load_async_to_lds_b128((as1_v4i_p)g, (as3_v4i_p)l, 0, 0);
    __builtin_amdgcn_global_load_async_to_lds_b128((as1_v4i_p)g, (as3_v4i_p)l, 16, 0);
#else
    *(bf16x8*)l       = *(const bf16x8*)g;
    *(bf16x8*)(l + 8) = *(const bf16x8*)(g + 8);
#endif
}
// copy 64 contiguous bytes (4 x b128)
static DEV void async_cp64B(const bf16_t* g, bf16_t* l) {
#if HAVE_ASYNC
    __builtin_amdgcn_global_load_async_to_lds_b128((as1_v4i_p)g, (as3_v4i_p)l, 0, 0);
    __builtin_amdgcn_global_load_async_to_lds_b128((as1_v4i_p)g, (as3_v4i_p)l, 16, 0);
    __builtin_amdgcn_global_load_async_to_lds_b128((as1_v4i_p)g, (as3_v4i_p)l, 32, 0);
    __builtin_amdgcn_global_load_async_to_lds_b128((as1_v4i_p)g, (as3_v4i_p)l, 48, 0);
#else
#pragma unroll
    for (int s = 0; s < 4; ++s)
        *(bf16x8*)(l + 8 * s) = *(const bf16x8*)(g + 8 * s);
#endif
}
static DEV void wait_async() {
#if HAVE_ASYNC
#if __has_builtin(__builtin_amdgcn_s_wait_asynccnt)
    __builtin_amdgcn_s_wait_asynccnt(0);
#else
    asm volatile("s_wait_asynccnt 0" ::: "memory");
#endif
#endif
}

static DEV f32x8 wmma_bf16(bf16x16 a, bf16x16 b, f32x8 c) {
    return __builtin_amdgcn_wmma_f32_16x16x32_bf16(false, a, false, b,
                                                   (short)0, c, false, false);
}

static DEV bf16x16 combine16(bf16x8 lo, bf16x8 hi) {
    return __builtin_shufflevector(lo, hi, 0, 1, 2, 3, 4, 5, 6, 7,
                                   8, 9, 10, 11, 12, 13, 14, 15);
}

static DEV bf16x8 ld_bf8(const bf16_t* p) { return *(const bf16x8*)p; }

static DEV bf16x8 pack8(float4 a, float4 b) {
    bf16x8 r;
    r[0] = (bf16_t)a.x; r[1] = (bf16_t)a.y; r[2] = (bf16_t)a.z; r[3] = (bf16_t)a.w;
    r[4] = (bf16_t)b.x; r[5] = (bf16_t)b.y; r[6] = (bf16_t)b.z; r[7] = (bf16_t)b.w;
    return r;
}
static DEV bf16x8 pack8s(float4 a, float4 b, float s) {
    bf16x8 r;
    r[0] = (bf16_t)(a.x * s); r[1] = (bf16_t)(a.y * s);
    r[2] = (bf16_t)(a.z * s); r[3] = (bf16_t)(a.w * s);
    r[4] = (bf16_t)(b.x * s); r[5] = (bf16_t)(b.y * s);
    r[6] = (bf16_t)(b.z * s); r[7] = (bf16_t)(b.w * s);
    return r;
}

// A-matrix 16x32 bf16 fragment: lane&15 = row M; lanes<16 hold
// K 0..15, lanes>=16 hold K 16..31. LDS layout [M][stride].
static DEV bf16x16 load_a_frag(const bf16_t* base, int stride, int m0, int k0, int lane) {
    const int row = m0 + (lane & 15);
    const int ka  = k0 + ((lane >> 4) << 4);
    return combine16(ld_bf8(base + row * stride + ka),
                     ld_bf8(base + row * stride + ka + 8));
}

// B-matrix 32x16 bf16 fragment: lane&15 = col N; lanes<16 hold K 0..15,
// lanes>=16 hold K 16..31. LDS layout transposed: [N][stride] contiguous in K.
static DEV bf16x16 load_b_frag(const bf16_t* base, int stride, int n0, int k0, int lane) {
    const int col = n0 + (lane & 15);
    const int kb  = k0 + ((lane >> 4) << 4);
    return combine16(ld_bf8(base + col * stride + kb),
                     ld_bf8(base + col * stride + kb + 8));
}

// ---------------------------------------------------------------------------
// Problem constants
// ---------------------------------------------------------------------------
static constexpr int NB     = 2;
static constexpr int L      = 2048;
static constexpr int DMODEL = 1024;
static constexpr int NHEADS = 16;
static constexpr int DV     = 64;
static constexpr int DFF    = 4096;
static constexpr int TOKENS = NB * L;  // 4096

// ---------------------------------------------------------------------------
// Pre-pass: straight f32 -> bf16 conversion (8 elements per thread)
// ---------------------------------------------------------------------------
__global__ __launch_bounds__(256)
void cvt_kernel(const float* __restrict__ in, bf16_t* __restrict__ out, int n8) {
    const int i = blockIdx.x * 256 + threadIdx.x;
    if (i < n8) {
        const float4* p = (const float4*)in + 2 * (size_t)i;
        *((bf16x8*)out + i) = pack8(p[0], p[1]);
    }
}

// ---------------------------------------------------------------------------
// Pre-pass: int mask -> additive f32 bias (0 or -inf)
// ---------------------------------------------------------------------------
__global__ __launch_bounds__(256)
void maskbias_kernel(const int* __restrict__ mask, float* __restrict__ mb, int n) {
    const int i = blockIdx.x * 256 + threadIdx.x;
    if (i < n) mb[i] = mask[i] ? 0.f : -__builtin_inff();
}

// ---------------------------------------------------------------------------
// Pre-pass: tiled transpose + f32 -> bf16.  in [R][C] f32 -> out [C][R] bf16.
// ---------------------------------------------------------------------------
__global__ __launch_bounds__(256)
void transpose_cvt_kernel(const float* __restrict__ in, bf16_t* __restrict__ out,
                          int R, int C, size_t in_batch, size_t out_batch) {
    __shared__ float tile[32][33];
    const float* ip = in + blockIdx.z * in_batch;
    bf16_t*      op = out + blockIdx.z * out_batch;
    const int r0 = blockIdx.y * 32, c0 = blockIdx.x * 32;
    const int tx = threadIdx.x & 31, ty = threadIdx.x >> 5;  // 32 x 8
#pragma unroll
    for (int j = 0; j < 32; j += 8)
        tile[ty + j][tx] = ip[(size_t)(r0 + ty + j) * C + c0 + tx];
    __syncthreads();
#pragma unroll
    for (int j = 0; j < 32; j += 8)
        op[(size_t)(c0 + ty + j) * R + r0 + tx] = (bf16_t)tile[tx][ty + j];
}

// ---------------------------------------------------------------------------
// Kernel 1: flash attention, transposed S (S^T = K Q^T, O^T = V^T P).
// 1/sqrt(d_model) folded into Q; mask folded into the S^T accumulator init.
// K/V staged by async global->LDS copies into double-buffered tiles.
// ---------------------------------------------------------------------------
__global__ __launch_bounds__(128)
void attn_kernel(const float* __restrict__ Q, const bf16_t* __restrict__ Kb,
                 const bf16_t* __restrict__ Vt, const float* __restrict__ mbias,
                 bf16_t* __restrict__ out) {
    __shared__ bf16_t Ks[2][32][72];   // [key][d] : A-frag source for K tile
    __shared__ bf16_t Vs[2][64][40];   // [d][key] : A-frag source for V^T tile

    const int tid  = threadIdx.x;
    const int lane = tid & 31;
    const int wave = tid >> 5;
    const int half = lane >> 4;
    const int cn   = lane & 15;        // query column this lane owns
    const int h    = blockIdx.y;
    const int n    = blockIdx.z;
    const int q_base = blockIdx.x * 64 + wave * 16;

    const float scale = 0.03125f;      // 1/sqrt(1024), exact in bf16
    const float NEG_INF = -__builtin_inff();

    // ---- persistent Q^T B-fragments, scale pre-folded ----
    const float* qp = Q + ((size_t)(n * L + q_base + cn)) * DMODEL + h * DV;
    bf16x16 qb[2];
#pragma unroll
    for (int c = 0; c < 2; ++c) {
        const int d0 = c * 32 + half * 16;
        float4 x0 = *(const float4*)(qp + d0);
        float4 x1 = *(const float4*)(qp + d0 + 4);
        float4 x2 = *(const float4*)(qp + d0 + 8);
        float4 x3 = *(const float4*)(qp + d0 + 12);
        qb[c] = combine16(pack8s(x0, x1, scale), pack8s(x2, x3, scale));
    }

    bf16x16 ones;                      // denominator A-tile
#pragma unroll
    for (int j = 0; j < 16; ++j) ones[j] = (bf16_t)1.0f;

    const f32x8 zero8 = {0.f, 0.f, 0.f, 0.f, 0.f, 0.f, 0.f, 0.f};
    f32x8 acc[4];
#pragma unroll
    for (int i = 0; i < 4; ++i) acc[i] = zero8;
    f32x8 accS = zero8;
    float mrun = NEG_INF;

    // staging thread mapping (pure contiguous copies)
    const int k_key = tid >> 2;          // 0..31
    const int k_d   = (tid & 3) * 16;    // 0,16,32,48
    const int v_d   = tid >> 1;          // 0..63
    const int v_kh  = (tid & 1) * 16;    // 0,16
    const float* mrow = mbias + n * L;

    auto issue = [&](int l0, int buf) {
        const bf16_t* kp = Kb + ((size_t)(n * L + l0 + k_key)) * DMODEL + h * DV + k_d;
        const bf16_t* vp = Vt + ((size_t)((n * NHEADS + h) * DV + v_d)) * L + l0 + v_kh;
        async_cp32B(kp, &Ks[buf][k_key][k_d]);
        async_cp32B(vp, &Vs[buf][v_d][v_kh]);
    };

    issue(0, 0);

    const int NIT = L / 32;
    for (int it = 0; it < NIT; ++it) {
        const int l0 = it * 32;
        wait_async();                   // slab it landed (this wave)
        __syncthreads();                // ... and in every other wave
        if (it + 1 < NIT) issue(l0 + 32, (it + 1) & 1);   // overlaps WMMAs
        const bf16_t* ksb = &Ks[it & 1][0][0];
        const bf16_t* vsb = &Vs[it & 1][0][0];

        // ---- mask bias IS the accumulator init (C rows are key indices) ----
        float4 m0a = *(const float4*)(mrow + l0 + 8 * half);
        float4 m0b = *(const float4*)(mrow + l0 + 8 * half + 4);
        float4 m1a = *(const float4*)(mrow + l0 + 16 + 8 * half);
        float4 m1b = *(const float4*)(mrow + l0 + 16 + 8 * half + 4);
        f32x8 st0 = {m0a.x, m0a.y, m0a.z, m0a.w, m0b.x, m0b.y, m0b.z, m0b.w};
        f32x8 st1 = {m1a.x, m1a.y, m1a.z, m1a.w, m1b.x, m1b.y, m1b.z, m1b.w};

        // ---- S^T = (K/32) Q^T + maskbias : two 16-key tiles, K-dim = 64 ----
        st0 = wmma_bf16(load_a_frag(ksb, 72, 0, 0, lane), qb[0], st0);
        st0 = wmma_bf16(load_a_frag(ksb, 72, 0, 32, lane), qb[1], st0);
        st1 = wmma_bf16(load_a_frag(ksb, 72, 16, 0, lane), qb[0], st1);
        st1 = wmma_bf16(load_a_frag(ksb, 72, 16, 32, lane), qb[1], st1);

        // ---- per-query running max: ONE cross-lane shuffle ----
        float cm = NEG_INF;
#pragma unroll
        for (int r = 0; r < 8; ++r) cm = fmaxf(cm, fmaxf(st0[r], st1[r]));
        cm = fmaxf(cm, __shfl_xor(cm, 16, 32));
        const float mnew  = fmaxf(mrun, cm);
        const float alpha = __expf(mrun - mnew);
        mrun = mnew;

        // ---- half-swap so each lane gets the 16 keys its B-frag needs ----
        float sw0[8], sw1[8];
#pragma unroll
        for (int r = 0; r < 8; ++r) {
            sw0[r] = __shfl_xor(st0[r], 16, 32);
            sw1[r] = __shfl_xor(st1[r], 16, 32);
        }
        bf16x16 pb;
#pragma unroll
        for (int r = 0; r < 8; ++r) {
            const float lo = half ? sw1[r] : st0[r];
            const float hi = half ? st1[r] : sw0[r];
            pb[r]     = (bf16_t)__expf(lo - mnew);
            pb[r + 8] = (bf16_t)__expf(hi - mnew);
        }

        // ---- rescale and accumulate O^T += V^T P, denom += 1^T P ----
#pragma unroll
        for (int mi = 0; mi < 4; ++mi) {
#pragma unroll
            for (int r = 0; r < 8; ++r) acc[mi][r] *= alpha;
            acc[mi] = wmma_bf16(load_a_frag(vsb, 40, mi * 16, 0, lane), pb, acc[mi]);
        }
#pragma unroll
        for (int r = 0; r < 8; ++r) accS[r] *= alpha;
        accS = wmma_bf16(ones, pb, accS);
    }

    // ---- normalize and store bf16: lane owns query column cn ----
    const float inv = 1.f / accS[0];
    bf16_t* outp = out + ((size_t)(n * L + q_base + cn)) * DMODEL + h * DV;
#pragma unroll
    for (int mi = 0; mi < 4; ++mi) {
        bf16x8 w;
#pragma unroll
        for (int r = 0; r < 8; ++r) w[r] = (bf16_t)(acc[mi][r] * inv);
        *(bf16x8*)(outp + mi * 16 + 8 * half) = w;
    }
}

// ---------------------------------------------------------------------------
// Kernel 2: 128x128-tile bf16 WMMA GEMM, f32 accumulate, k-slab 64.
// A bf16 [M][K] row-major; Bt bf16 [N][K] (pre-transposed weights).
// Async global->LDS staging, double-buffered, one barrier per slab.
//   EPI 0: f32 out = A@B + bias + resid      EPI 1: bf16 out = relu(A@B + bias)
// ---------------------------------------------------------------------------
template <int EPI>
__global__ __launch_bounds__(256)
void gemm_kernel(const bf16_t* __restrict__ A, const bf16_t* __restrict__ Bt,
                 const float* __restrict__ bias, const float* __restrict__ resid,
                 void* __restrict__ outv, int M, int N, int K) {
    __shared__ bf16_t As[2][128][72];   // [m][k]
    __shared__ bf16_t Bs[2][128][72];   // [n][k]

    const int tid  = threadIdx.x;
    const int lane = tid & 31;
    const int wave = tid >> 5;
    const int wm   = wave >> 1;            // 0..3 -> 32-row slab
    const int wn   = wave & 1;             // 0..1 -> 64-col slab
    const int m_blk = blockIdx.y * 128;
    const int n_blk = blockIdx.x * 128;

    f32x8 acc[2][4];
    const f32x8 zero8 = {0.f, 0.f, 0.f, 0.f, 0.f, 0.f, 0.f, 0.f};
#pragma unroll
    for (int mi = 0; mi < 2; ++mi)
#pragma unroll
        for (int ni = 0; ni < 4; ++ni) acc[mi][ni] = zero8;

    const int c_row = tid >> 1;            // 0..127
    const int c_k   = (tid & 1) * 32;      // 0,32

    auto issue = [&](int k0, int buf) {
        const bf16_t* ap = A  + (size_t)(m_blk + c_row) * K + k0 + c_k;
        const bf16_t* bp = Bt + (size_t)(n_blk + c_row) * K + k0 + c_k;
        __builtin_prefetch(ap + 64, 0, 1);      // L2 stays one more slab ahead
        __builtin_prefetch(bp + 64, 0, 1);
        async_cp64B(ap, &As[buf][c_row][c_k]);
        async_cp64B(bp, &Bs[buf][c_row][c_k]);
    };

    issue(0, 0);

    const int NIT = K >> 6;
    for (int it = 0; it < NIT; ++it) {
        wait_async();                    // slab it landed in LDS (this wave)
        __syncthreads();                 // ... and in every other wave
        if (it + 1 < NIT) issue((it + 1) << 6, (it + 1) & 1);  // overlaps WMMAs
        const bf16_t* as = &As[it & 1][0][0];
        const bf16_t* bs = &Bs[it & 1][0][0];
#pragma unroll
        for (int kc = 0; kc < 2; ++kc) {
            bf16x16 af[2], bfrag[4];
#pragma unroll
            for (int mi = 0; mi < 2; ++mi)
                af[mi] = load_a_frag(as, 72, wm * 32 + mi * 16, kc * 32, lane);
#pragma unroll
            for (int ni = 0; ni < 4; ++ni)
                bfrag[ni] = load_b_frag(bs, 72, wn * 64 + ni * 16, kc * 32, lane);
#pragma unroll
            for (int mi = 0; mi < 2; ++mi)
#pragma unroll
                for (int ni = 0; ni < 4; ++ni)
                    acc[mi][ni] = wmma_bf16(af[mi], bfrag[ni], acc[mi][ni]);
        }
    }

    // epilogue (C-layout: row = r + 8*(lane>=16), col = lane&15)
    const int half = lane >> 4;
    const int cn   = lane & 15;
#pragma unroll
    for (int mi = 0; mi < 2; ++mi)
#pragma unroll
        for (int ni = 0; ni < 4; ++ni)
#pragma unroll
            for (int r = 0; r < 8; ++r) {
                const int row = m_blk + wm * 32 + mi * 16 + r + 8 * half;
                const int col = n_blk + wn * 64 + ni * 16 + cn;
                float v = acc[mi][ni][r] + bias[col];
                if (EPI == 1) {
                    ((bf16_t*)outv)[(size_t)row * N + col] = (bf16_t)fmaxf(v, 0.f);
                } else {
                    v += resid[(size_t)row * N + col];
                    ((float*)outv)[(size_t)row * N + col] = v;
                }
            }
}

// ---------------------------------------------------------------------------
// Kernel 3: LayerNorm over 1024-wide rows; optionally also emits bf16 copy
// (the bf16 copy feeds the next GEMM's A operand).
// ---------------------------------------------------------------------------
template <bool BF16OUT>
__global__ __launch_bounds__(256)
void ln_kernel(const float* __restrict__ x, const float* __restrict__ g,
               const float* __restrict__ b, float* __restrict__ y,
               bf16_t* __restrict__ yb) {
    const int row = blockIdx.x;
    const int t   = threadIdx.x;
    const float4 v = ((const float4*)(x + (size_t)row * DMODEL))[t];
    float s  = v.x + v.y + v.z + v.w;
    float s2 = v.x * v.x + v.y * v.y + v.z * v.z + v.w * v.w;
#pragma unroll
    for (int m = 16; m >= 1; m >>= 1) {
        s  += __shfl_xor(s, m, 32);
        s2 += __shfl_xor(s2, m, 32);
    }
    __shared__ float rs[8], rs2[8];
    const int lane = t & 31, wave = t >> 5;
    if (lane == 0) { rs[wave] = s; rs2[wave] = s2; }
    __syncthreads();
    float ts = 0.f, ts2 = 0.f;
#pragma unroll
    for (int i = 0; i < 8; ++i) { ts += rs[i]; ts2 += rs2[i]; }
    const float mu   = ts * (1.f / DMODEL);
    const float var  = ts2 * (1.f / DMODEL) - mu * mu;
    const float rstd = rsqrtf(var + 1e-5f);
    const int c = t * 4;
    float4 o;
    o.x = (v.x - mu) * rstd * g[c + 0] + b[c + 0];
    o.y = (v.y - mu) * rstd * g[c + 1] + b[c + 1];
    o.z = (v.z - mu) * rstd * g[c + 2] + b[c + 2];
    o.w = (v.w - mu) * rstd * g[c + 3] + b[c + 3];
    ((float4*)(y + (size_t)row * DMODEL))[t] = o;
    if (BF16OUT) {
        bf16x4 ob;
        ob[0] = (bf16_t)o.x; ob[1] = (bf16_t)o.y;
        ob[2] = (bf16_t)o.z; ob[3] = (bf16_t)o.w;
        *(bf16x4*)(yb + (size_t)row * DMODEL + c) = ob;
    }
}

// ---------------------------------------------------------------------------
// Launch
// ---------------------------------------------------------------------------
extern "C" void kernel_launch(void* const* d_in, const int* in_sizes, int n_in,
                              void* d_out, int out_size, void* d_ws, size_t ws_size,
                              hipStream_t stream) {
    const float* q     = (const float*)d_in[0];
    const float* k     = (const float*)d_in[1];
    const float* v     = (const float*)d_in[2];
    const int*   mask  = (const int*)d_in[3];
    const float* fc_w  = (const float*)d_in[4];
    const float* fc_b  = (const float*)d_in[5];
    const float* ln1_g = (const float*)d_in[6];
    const float* ln1_b = (const float*)d_in[7];
    const float* ff_w1 = (const float*)d_in[8];
    const float* ff_b1 = (const float*)d_in[9];
    const float* ff_w2 = (const float*)d_in[10];
    const float* ff_b2 = (const float*)d_in[11];
    const float* ln2_g = (const float*)d_in[12];
    const float* ln2_b = (const float*)d_in[13];
    float* out = (float*)d_out;

    // ---- workspace carve-up (256B aligned) ----
    char* wsp = (char*)d_ws;
    auto alloc = [&](size_t bytes) -> char* {
        char* p = wsp;
        wsp += (bytes + 255) & ~(size_t)255;
        return p;
    };
    const size_t TD = (size_t)TOKENS * DMODEL;
    bf16_t* fcw_t  = (bf16_t*)alloc((size_t)DMODEL * DMODEL * 2);  // [N][K]
    bf16_t* ffw1_t = (bf16_t*)alloc((size_t)DFF * DMODEL * 2);     // [4096][1024]
    bf16_t* ffw2_t = (bf16_t*)alloc((size_t)DMODEL * DFF * 2);     // [1024][4096]
    bf16_t* Kb     = (bf16_t*)alloc(TD * 2);                       // [n][l][1024]
    bf16_t* Vt     = (bf16_t*)alloc(TD * 2);                       // [n][h][d][l]
    bf16_t* attnb  = (bf16_t*)alloc(TD * 2);                       // attention out
    bf16_t* x1b    = (bf16_t*)alloc(TD * 2);                       // LN1 out bf16
    bf16_t* hb     = (bf16_t*)alloc((size_t)TOKENS * DFF * 2);     // FFN hidden
    float*  buf1   = (float*)alloc(TD * 4);                        // pre-LN scratch
    float*  buf2   = (float*)alloc(TD * 4);                        // x1 f32 (resid)
    float*  mbias  = (float*)alloc((size_t)NB * L * 4);            // mask bias

    // ---- pre-pass: bf16 conversions / transposes / mask bias ----
    cvt_kernel<<<(int)(TD / 8 / 256), 256, 0, stream>>>(k, Kb, (int)(TD / 8));
    maskbias_kernel<<<(NB * L) / 256, 256, 0, stream>>>(mask, mbias, NB * L);
    transpose_cvt_kernel<<<dim3(DMODEL / 32, L / 32, NB), 256, 0, stream>>>(
        v, Vt, L, DMODEL, (size_t)L * DMODEL, (size_t)DMODEL * L);
    transpose_cvt_kernel<<<dim3(DMODEL / 32, DMODEL / 32, 1), 256, 0, stream>>>(
        fc_w, fcw_t, DMODEL, DMODEL, 0, 0);
    transpose_cvt_kernel<<<dim3(DFF / 32, DMODEL / 32, 1), 256, 0, stream>>>(
        ff_w1, ffw1_t, DMODEL, DFF, 0, 0);
    transpose_cvt_kernel<<<dim3(DMODEL / 32, DFF / 32, 1), 256, 0, stream>>>(
        ff_w2, ffw2_t, DFF, DMODEL, 0, 0);

    // 1. attention -> attnb (bf16)
    attn_kernel<<<dim3(L / 64, NHEADS, NB), 128, 0, stream>>>(q, Kb, Vt, mbias, attnb);
    // 2. fc: attnb @ fc_w + fc_b + queries -> buf1 (f32)
    gemm_kernel<0><<<dim3(DMODEL / 128, TOKENS / 128), 256, 0, stream>>>(
        attnb, fcw_t, fc_b, q, buf1, TOKENS, DMODEL, DMODEL);
    // 3. LN1 -> buf2 (f32 resid) + x1b (bf16 GEMM A)
    ln_kernel<true><<<TOKENS, 256, 0, stream>>>(buf1, ln1_g, ln1_b, buf2, x1b);
    // 4. ff1 + relu: x1b @ ff_w1 -> hb (bf16)
    gemm_kernel<1><<<dim3(DFF / 128, TOKENS / 128), 256, 0, stream>>>(
        x1b, ffw1_t, ff_b1, nullptr, hb, TOKENS, DFF, DMODEL);
    // 5. ff2 + residual(x1): hb @ ff_w2 + buf2 -> buf1 (f32)
    gemm_kernel<0><<<dim3(DMODEL / 128, TOKENS / 128), 256, 0, stream>>>(
        hb, ffw2_t, ff_b2, buf2, buf1, TOKENS, DMODEL, DFF);
    // 6. LN2 -> d_out
    ln_kernel<false><<<TOKENS, 256, 0, stream>>>(buf1, ln2_g, ln2_b, out, nullptr);
}
